// MoEBlock_36962488549729
// MI455X (gfx1250) — matrix operand order repaired and checked by reading, not compiled, not run on hardware
//
#include <hip/hip_runtime.h>
#include <hip/hip_bf16.h>

#define N_TOK 8192
#define DIM   1024
#define HID   4096
#define NE    8
#define LN_EPS 1e-5f

typedef __attribute__((ext_vector_type(8)))  float  v8f;
typedef __attribute__((ext_vector_type(8)))  __bf16 v8bf;
typedef __attribute__((ext_vector_type(16))) __bf16 v16bf;

// ---------------------------------------------------------------------------
// 1) fp32 -> bf16 convert (grid-stride, 4 elems/thread)
// ---------------------------------------------------------------------------
__global__ void f2bf_kernel(const float* __restrict__ in, __bf16* __restrict__ out, int n4) {
  for (int i = blockIdx.x * blockDim.x + threadIdx.x; i < n4; i += gridDim.x * blockDim.x) {
    float4 v = ((const float4*)in)[i];
    __bf16 o[4] = {(__bf16)v.x, (__bf16)v.y, (__bf16)v.z, (__bf16)v.w};
    *(uint2*)&out[4 * i] = *(const uint2*)o;
  }
}

// ---------------------------------------------------------------------------
// 2) zero per-expert counters
// ---------------------------------------------------------------------------
__global__ void zero_counts_kernel(int* counts, int* run) {
  if (threadIdx.x < NE) { counts[threadIdx.x] = 0; run[threadIdx.x] = 0; }
}

// ---------------------------------------------------------------------------
// 3) gating: one wave32 per token. logits = x @ gate_w^T + gate_b,
//    softmax over 8, top-2, renormalizing softmax over the 2 scores.
// ---------------------------------------------------------------------------
__global__ void gate_kernel(const float* __restrict__ x,
                            const float* __restrict__ gw,
                            const float* __restrict__ gb,
                            int* __restrict__ topk_idx,
                            float* __restrict__ topk_w,
                            int* __restrict__ counts) {
  const int wave = threadIdx.x >> 5;
  const int lane = threadIdx.x & 31;
  const int tok  = blockIdx.x * 8 + wave;
  if (tok >= N_TOK) return;

  float s[NE];
#pragma unroll
  for (int e = 0; e < NE; ++e) s[e] = 0.f;

  const float* xr = x + (size_t)tok * DIM;
  for (int j = lane; j < DIM; j += 32) {
    float xv = xr[j];
#pragma unroll
    for (int e = 0; e < NE; ++e) s[e] += xv * gw[e * DIM + j];
  }
#pragma unroll
  for (int e = 0; e < NE; ++e) {
#pragma unroll
    for (int m = 16; m >= 1; m >>= 1) s[e] += __shfl_xor(s[e], m, 32);
  }

  if (lane == 0) {
    float lg[NE];
    float mx = -1e30f;
#pragma unroll
    for (int e = 0; e < NE; ++e) { lg[e] = s[e] + gb[e]; mx = fmaxf(mx, lg[e]); }
    float sum = 0.f;
#pragma unroll
    for (int e = 0; e < NE; ++e) { lg[e] = __expf(lg[e] - mx); sum += lg[e]; }
    float inv = 1.f / sum;
#pragma unroll
    for (int e = 0; e < NE; ++e) lg[e] *= inv;

    int i0 = 0;
#pragma unroll
    for (int e = 1; e < NE; ++e) if (lg[e] > lg[i0]) i0 = e;
    int i1 = (i0 == 0) ? 1 : 0;
#pragma unroll
    for (int e = 0; e < NE; ++e) if (e != i0 && lg[e] > lg[i1]) i1 = e;

    float e1 = __expf(lg[i1] - lg[i0]);   // s0 >= s1
    float d  = 1.f / (1.f + e1);
    topk_idx[2 * tok + 0] = i0;
    topk_idx[2 * tok + 1] = i1;
    topk_w[2 * tok + 0] = d;
    topk_w[2 * tok + 1] = e1 * d;
    atomicAdd(&counts[i0], 1);
    atomicAdd(&counts[i1], 1);
  }
}

// ---------------------------------------------------------------------------
// 4) exclusive scan over 8 expert counts
// ---------------------------------------------------------------------------
__global__ void scan_kernel(const int* __restrict__ counts, int* __restrict__ offsets) {
  if (threadIdx.x == 0 && blockIdx.x == 0) {
    int acc = 0;
    for (int e = 0; e < NE; ++e) { offsets[e] = acc; acc += counts[e]; }
  }
}

// ---------------------------------------------------------------------------
// 5) build compacted routing
// ---------------------------------------------------------------------------
__global__ void fill_kernel(const int* __restrict__ topk_idx,
                            const int* __restrict__ offsets,
                            int* __restrict__ run,
                            int* __restrict__ tok_of_row,
                            int* __restrict__ row_of) {
  int id = blockIdx.x * blockDim.x + threadIdx.x;
  if (id >= 2 * N_TOK) return;
  int t = id >> 1;
  int e = topk_idx[id];
  int p = atomicAdd(&run[e], 1);
  int row = offsets[e] + p;
  tok_of_row[row] = t;
  row_of[id] = row;
}

// ---------------------------------------------------------------------------
// 6/7) tiled WMMA GEMM with async global->LDS double buffering.
//   OUT[rows_e x Nc] = act(A x W_e^T + bias_e)
//   Block tile 128x128, K-chunk 64; 8 waves each own a 64x32 wave tile
//   (8 f32 accumulator fragments, 16 WMMAs per K-chunk).
//   Tiles stream via GLOBAL_LOAD_ASYNC_TO_LDS_B128 (ASYNCcnt); per K-half all
//   12 fragment ds_loads are issued before the 8 back-to-back WMMAs.
// ---------------------------------------------------------------------------
#define LDS_STRIDE 72                 // bf16 elems per row (144B, 16B aligned)
#define TILE_ELEMS (128 * LDS_STRIDE)
#define TILE_BYTES (TILE_ELEMS * 2)

template <bool GATHER, bool RELU_BF16>
__global__ void moe_gemm_kernel(const __bf16* __restrict__ A, int lda,
                                const int* __restrict__ tok_of_row,
                                const __bf16* __restrict__ Wbase, size_t wstride,
                                const float* __restrict__ biasBase, int biasStride,
                                void* __restrict__ outBase, int ldo,
                                const int* __restrict__ counts,
                                const int* __restrict__ offsets,
                                int K) {
  const int e     = blockIdx.z;
  const int rows  = counts[e];
  const int row0  = blockIdx.y * 128;
  if (row0 >= rows) return;                  // uniform per block: EXEC stays full
  const int off_e = offsets[e];
  const int n0    = blockIdx.x * 128;

  const __bf16* W    = Wbase + (size_t)e * wstride;
  const float*  bias = biasBase + (size_t)e * biasStride;

  const int tid   = threadIdx.x;
  const int wave  = tid >> 5;
  const int lane  = tid & 31;
  const int wm    = wave & 1;                // 2 waves along M  (64 rows each)
  const int wn    = wave >> 1;               // 4 waves along N  (32 cols each)
  const int lhalf = lane >> 4;               // contiguous K-half select (k = 16*lhalf + s)
  const int l16   = lane & 15;

  __shared__ __align__(16) __bf16 As[2 * TILE_ELEMS];
  __shared__ __align__(16) __bf16 Bs[2 * TILE_ELEMS];

  // ---- hoisted per-thread staging pointers: thread stages 16B chunk
  //      (row rbase+32j, 8*cchunk..8*cchunk+7) of each 128x64 tile ----
  const int cchunk = tid & 7;
  const int rbase  = tid >> 3;
  const __bf16* aptr[4];
  const __bf16* bptr[4];
  uint32_t aoff[4], boff[4];
  const uint32_t As0 = (uint32_t)(uintptr_t)As;
  const uint32_t Bs0 = (uint32_t)(uintptr_t)Bs;
#pragma unroll
  for (int j = 0; j < 4; ++j) {
    int r  = rbase + 32 * j;
    int ar = row0 + r;
    if (ar > rows - 1) ar = rows - 1;        // clamp: junk rows never stored
    int src = off_e + ar;
    if (GATHER) src = tok_of_row[src];
    aptr[j] = A + (size_t)src * lda + cchunk * 8;
    bptr[j] = W + (size_t)(n0 + r) * K + cchunk * 8;
    aoff[j] = As0 + r * (LDS_STRIDE * 2) + cchunk * 16;
    boff[j] = Bs0 + r * (LDS_STRIDE * 2) + cchunk * 16;
  }

  auto stage = [&](int k0, int buf) {
    uint32_t bb = (uint32_t)buf * TILE_BYTES;
#pragma unroll
    for (int j = 0; j < 4; ++j) {
      unsigned long long ga = (unsigned long long)(uintptr_t)(aptr[j] + k0);
      asm volatile("global_load_async_to_lds_b128 %0, %1, off"
                   :: "v"(aoff[j] + bb), "v"(ga) : "memory");
    }
#pragma unroll
    for (int j = 0; j < 4; ++j) {
      unsigned long long gb = (unsigned long long)(uintptr_t)(bptr[j] + k0);
      asm volatile("global_load_async_to_lds_b128 %0, %1, off"
                   :: "v"(boff[j] + bb), "v"(gb) : "memory");
    }
  };

  v8f acc[4][2] = {};

  stage(0, 0);                               // prologue: tile 0 in flight
  const int nk = K >> 6;                     // K / 64
  for (int kt = 0; kt < nk; ++kt) {
    const int buf = kt & 1;
    if (kt + 1 < nk) {
      stage((kt + 1) << 6, buf ^ 1);         // next tile in flight (8 ops/wave)
      asm volatile("s_wait_asynccnt 8" ::: "memory");   // current tile landed
    } else {
      asm volatile("s_wait_asynccnt 0" ::: "memory");
    }
    __syncthreads();                         // all waves' tiles visible

    const __bf16* Ab = As + buf * TILE_ELEMS;
    const __bf16* Bb = Bs + buf * TILE_ELEMS;
#pragma unroll
    for (int ki = 0; ki < 2; ++ki) {
      const int k_in = ki * 32;
      // ---- issue ALL fragment loads first (12x ds_load_b128) ----
      v16bf bfrag[2];
      v8bf  alo[4], ahi[4];
#pragma unroll
      for (int t = 0; t < 2; ++t)
        bfrag[t] = *(const v16bf*)&Bb[(wn * 32 + t * 16 + l16) * LDS_STRIDE +
                                      k_in + lhalf * 16];
#pragma unroll
      for (int s = 0; s < 4; ++s) {
        const __bf16* ap = &Ab[(wm * 64 + s * 16 + l16) * LDS_STRIDE +
                               k_in + lhalf * 16];
        alo[s] = *(const v8bf*)ap;           // K 0-7  (16-23 upper half)
        ahi[s] = *(const v8bf*)(ap + 8);     // K 8-15 (24-31 upper half)
      }
      // ---- then 8 independent WMMAs back-to-back ----
#pragma unroll
      for (int s = 0; s < 4; ++s) {
        v16bf afrag = __builtin_shufflevector(alo[s], ahi[s],
                                              0, 1, 2, 3, 4, 5, 6, 7,
                                              8, 9, 10, 11, 12, 13, 14, 15);
#pragma unroll
        for (int t = 0; t < 2; ++t)
          acc[s][t] = __builtin_amdgcn_wmma_f32_16x16x32_bf16(
              false, afrag, false, bfrag[t], (short)0, acc[s][t], false, false);
      }
    }
    __syncthreads();                         // tile fully consumed, reusable
  }

  // ---- epilogue: bias (+relu) and store ----
#pragma unroll
  for (int s = 0; s < 4; ++s) {
#pragma unroll
    for (int t = 0; t < 2; ++t) {
      int nc = n0 + wn * 32 + t * 16 + l16;
      float bv = bias[nc];
      int mbase = row0 + wm * 64 + s * 16 + lhalf * 8;
#pragma unroll
      for (int v = 0; v < 8; ++v) {
        int m = mbase + v;
        if (m < rows) {
          size_t orow = (size_t)(off_e + m);
          float val = acc[s][t][v] + bv;
          if (RELU_BF16) {
            ((__bf16*)outBase)[orow * ldo + nc] = (__bf16)fmaxf(val, 0.f);
          } else {
            ((float*)outBase)[orow * ldo + nc] = val;
          }
        }
      }
    }
  }
}

// ---------------------------------------------------------------------------
// 8) combine: out = LN(x + w0*y[r0] + w1*y[r1]) ; one block per token
// ---------------------------------------------------------------------------
__global__ void combine_kernel(const float* __restrict__ x,
                               const float* __restrict__ y,
                               const int* __restrict__ row_of,
                               const float* __restrict__ tw,
                               const float* __restrict__ lnw,
                               const float* __restrict__ lnb,
                               float* __restrict__ out) {
  const int t   = blockIdx.x;
  const int tid = threadIdx.x;
  const int r0 = row_of[2 * t + 0], r1 = row_of[2 * t + 1];
  const float w0 = tw[2 * t + 0], w1 = tw[2 * t + 1];

  __shared__ float red[256];
  __shared__ float s_mu, s_rstd;

  float v[4];
  float lsum = 0.f;
#pragma unroll
  for (int k = 0; k < 4; ++k) {
    int c = tid + k * 256;
    float r = x[(size_t)t * DIM + c]
            + w0 * y[(size_t)r0 * DIM + c]
            + w1 * y[(size_t)r1 * DIM + c];
    v[k] = r;
    lsum += r;
  }
  red[tid] = lsum; __syncthreads();
  for (int s = 128; s > 0; s >>= 1) { if (tid < s) red[tid] += red[tid + s]; __syncthreads(); }
  if (tid == 0) s_mu = red[0] * (1.f / DIM);
  __syncthreads();
  float mu = s_mu;

  float lvar = 0.f;
#pragma unroll
  for (int k = 0; k < 4; ++k) { float d = v[k] - mu; lvar += d * d; }
  red[tid] = lvar; __syncthreads();
  for (int s = 128; s > 0; s >>= 1) { if (tid < s) red[tid] += red[tid + s]; __syncthreads(); }
  if (tid == 0) s_rstd = rsqrtf(red[0] * (1.f / DIM) + LN_EPS);
  __syncthreads();
  float rstd = s_rstd;

#pragma unroll
  for (int k = 0; k < 4; ++k) {
    int c = tid + k * 256;
    out[(size_t)t * DIM + c] = (v[k] - mu) * rstd * lnw[c] + lnb[c];
  }
}

// ---------------------------------------------------------------------------
// launcher
// ---------------------------------------------------------------------------
extern "C" void kernel_launch(void* const* d_in, const int* in_sizes, int n_in,
                              void* d_out, int out_size, void* d_ws, size_t ws_size,
                              hipStream_t stream) {
  const float* x    = (const float*)d_in[0];
  const float* gw   = (const float*)d_in[1];
  const float* gb   = (const float*)d_in[2];
  const float* w1   = (const float*)d_in[3];
  const float* b1   = (const float*)d_in[4];
  const float* w2   = (const float*)d_in[5];
  const float* b2   = (const float*)d_in[6];
  const float* lnw  = (const float*)d_in[7];
  const float* lnb  = (const float*)d_in[8];
  float* out = (float*)d_out;

  // ---- carve workspace ----
  char* ws = (char*)d_ws;
  size_t off = 0;
  auto carve = [&](size_t bytes) { char* p = ws + off; off = (off + bytes + 255) & ~(size_t)255; return p; };
  __bf16* xb   = (__bf16*)carve((size_t)N_TOK * DIM * 2);
  __bf16* w1b  = (__bf16*)carve((size_t)NE * HID * DIM * 2);
  __bf16* w2b  = (__bf16*)carve((size_t)NE * DIM * HID * 2);
  __bf16* h    = (__bf16*)carve((size_t)2 * N_TOK * HID * 2);
  float*  yrow = (float*) carve((size_t)2 * N_TOK * DIM * 4);
  int*   topk_idx = (int*)  carve((size_t)2 * N_TOK * 4);
  float* topk_w   = (float*)carve((size_t)2 * N_TOK * 4);
  int*   counts   = (int*)  carve(NE * 4);
  int*   offsets  = (int*)  carve(NE * 4);
  int*   run      = (int*)  carve(NE * 4);
  int*   tok_of_row = (int*)carve((size_t)2 * N_TOK * 4);
  int*   row_of     = (int*)carve((size_t)2 * N_TOK * 4);
  (void)ws_size; (void)n_in; (void)in_sizes; (void)out_size;

  // 1) bf16 conversions
  f2bf_kernel<<<2048, 256, 0, stream>>>(x,  xb,  N_TOK * DIM / 4);
  f2bf_kernel<<<8192, 256, 0, stream>>>(w1, w1b, NE * HID * DIM / 4);
  f2bf_kernel<<<8192, 256, 0, stream>>>(w2, w2b, NE * DIM * HID / 4);

  // 2) routing
  zero_counts_kernel<<<1, 32, 0, stream>>>(counts, run);
  gate_kernel<<<N_TOK / 8, 256, 0, stream>>>(x, gw, gb, topk_idx, topk_w, counts);
  scan_kernel<<<1, 32, 0, stream>>>(counts, offsets);
  fill_kernel<<<(2 * N_TOK) / 256, 256, 0, stream>>>(topk_idx, offsets, run, tok_of_row, row_of);

  // 3) FFN pass 1: h = relu(x_gathered @ w1^T + b1)  -> bf16 [2N, HID]
  {
    dim3 grid(HID / 128, N_TOK / 128, NE);
    moe_gemm_kernel<true, true><<<grid, 256, 0, stream>>>(
        xb, DIM, tok_of_row, w1b, (size_t)HID * DIM, b1, HID,
        (void*)h, HID, counts, offsets, DIM);
  }
  // 4) FFN pass 2: y = h @ w2^T + b2 -> fp32 [2N, DIM]
  {
    dim3 grid(DIM / 128, N_TOK / 128, NE);
    moe_gemm_kernel<false, false><<<grid, 256, 0, stream>>>(
        h, HID, nullptr, w2b, (size_t)DIM * HID, b2, DIM,
        (void*)yrow, DIM, counts, offsets, HID);
  }

  // 5) weighted combine + residual + LayerNorm
  combine_kernel<<<N_TOK, 256, 0, stream>>>(x, yrow, row_of, topk_w, lnw, lnb, out);
}
